// GTLayer_34540126994678
// MI455X (gfx1250) — matrix operand order repaired and checked by reading, not compiled, not run on hardware
//
#include <hip/hip_runtime.h>

// ---------------- CDNA5 types ----------------
typedef __bf16 bf16_t;
typedef __attribute__((ext_vector_type(16))) __bf16 v16bf;
typedef __attribute__((ext_vector_type(8)))  float  v8f;

#define NN   50000LL
#define EE   800000LL
#define HDIM 128
#define NHEAD 8

// monotone float<->uint encoding for atomicMax on floats
__device__ __forceinline__ unsigned fenc(float f){
  unsigned b = __float_as_uint(f);
  return (b & 0x80000000u) ? ~b : (b | 0x80000000u);
}
__device__ __forceinline__ float fdec(unsigned u){
  unsigned b = (u & 0x80000000u) ? (u & 0x7fffffffu) : ~u;
  return __uint_as_float(b);
}

// ---------------- tiny utility kernels ----------------
__global__ void fill_f32(float* p, long long n){
  long long i = (long long)blockIdx.x * blockDim.x + threadIdx.x;
  long long st = (long long)gridDim.x * blockDim.x;
  for (; i < n; i += st) p[i] = 0.f;
}
__global__ void fill_u32(unsigned* p, int n){
  int i = threadIdx.x;
  if (i < n) p[i] = 0u;
}
__global__ void cvt_bf16(const float* a, bf16_t* b, long long n){
  long long i = (long long)blockIdx.x * blockDim.x + threadIdx.x;
  long long st = (long long)gridDim.x * blockDim.x;
  for (; i < n; i += st) b[i] = (bf16_t)a[i];
}
// W [K][Nc] f32 row-major -> BT [Nc][K] bf16 (so B columns are contiguous)
__global__ void transpose_bf16(const float* W, bf16_t* BT, int K, int Nc){
  long long tot = (long long)K * Nc;
  long long i = (long long)blockIdx.x * blockDim.x + threadIdx.x;
  long long st = (long long)gridDim.x * blockDim.x;
  for (; i < tot; i += st){
    int k = (int)(i / Nc), n = (int)(i % Nc);
    BT[(long long)n * K + k] = (bf16_t)W[i];
  }
}

// ---------------- WMMA fragment loaders ----------------
// A 16x32 bf16 frag: lane<16 -> M=lane, K = k0+0..7 & k0+16..23
//                    lane>=16 -> M=lane-16, K = k0+8..15 & k0+24..31
__device__ __forceinline__ v16bf load_a_frag(const bf16_t* A, int lda, int m0, int k0, int lane){
  int r  = lane & 15;
  int kb = k0 + ((lane & 16) ? 8 : 0);
  const bf16_t* p = A + (long long)(m0 + r) * lda + kb;
  union { uint4 u; bf16_t h[8]; } u0, u1;
  u0.u = *(const uint4*)p;
  u1.u = *(const uint4*)(p + 16);
  v16bf f;
#pragma unroll
  for (int i = 0; i < 8; ++i){ f[i] = u0.h[i]; f[i + 8] = u1.h[i]; }
  return f;
}
// B 32x16 bf16 frag from BT[Nc][K]: lane<16 -> N=n0+lane, K=k0+0..7 & k0+16..23
//                                   lane>=16 -> N=n0+lane-16, K=k0+8..15 & k0+24..31
__device__ __forceinline__ v16bf load_b_frag(const bf16_t* BT, int ldb, int n0, int k0, int lane){
  int c  = lane & 15;
  int kb = k0 + ((lane & 16) ? 8 : 0);
  const bf16_t* p = BT + (long long)(n0 + c) * ldb + kb;
  union { uint4 u; bf16_t h[8]; } u0, u1;
  u0.u = *(const uint4*)p;
  u1.u = *(const uint4*)(p + 16);
  v16bf f;
#pragma unroll
  for (int i = 0; i < 8; ++i){ f[i] = u0.h[i]; f[i + 8] = u1.h[i]; }
  return f;
}

// ---------------- generic bf16 WMMA GEMM (software-pipelined K loop) ----------------
// out = act(A[M,K] @ BT^T + bias);  A bf16 row-major, BT bf16 [Nc][K]
// act: 0 = none, 1 = exact GELU.  outf / outb may be null.
__global__ void gemm_bf16(const bf16_t* __restrict__ A, const bf16_t* __restrict__ BT,
                          const float* __restrict__ bias,
                          float* __restrict__ outf, bf16_t* __restrict__ outb,
                          int M, int K, int Nc, int act){
  int lane = threadIdx.x & 31;
  int wave = threadIdx.x >> 5;
  int m0 = blockIdx.x * 16;
  if (m0 >= M) return;
  for (int nt = wave; nt < (Nc >> 4); nt += 8){
    int n0 = nt << 4;
    v8f c = {};
    // double-buffered pipeline: fetch k+32 fragments before the k WMMA so the
    // scheduler can wait on older loads only (loadcnt<=4) instead of draining.
    v16bf a = load_a_frag(A, K, m0, 0, lane);
    v16bf b = load_b_frag(BT, K, n0, 0, lane);
    int k0 = 0;
    for (; k0 + 32 < K; k0 += 32){
      v16bf an = load_a_frag(A, K, m0, k0 + 32, lane);
      v16bf bn = load_b_frag(BT, K, n0, k0 + 32, lane);
      c = __builtin_amdgcn_wmma_f32_16x16x32_bf16(false, a, false, b, (short)0, c, false, false);
      a = an; b = bn;
    }
    c = __builtin_amdgcn_wmma_f32_16x16x32_bf16(false, a, false, b, (short)0, c, false, false);
    int col = n0 + (lane & 15);
    float bi = bias ? bias[col] : 0.f;
#pragma unroll
    for (int r = 0; r < 8; ++r){
      int row = m0 + ((lane & 16) ? (8 + r) : r);
      float v = c[r] + bi;
      if (act) v = 0.5f * v * (1.f + erff(v * 0.70710678118654752f));
      long long idx = (long long)row * Nc + col;
      if (outf) outf[idx] = v;
      if (outb) outb[idx] = (bf16_t)v;
    }
  }
}

// ---------------- edge attention logits + global per-head max ----------------
// attn[e,h] = dot(Xq[dst]_h, Xk[src]_h)/4 + (e @ We)[e,h]
__global__ void edge_logits(const float* __restrict__ Xq, const float* __restrict__ Xk,
                            const float* __restrict__ efeat, const float* __restrict__ We,
                            const int* __restrict__ src, const int* __restrict__ dst,
                            float* __restrict__ attn, unsigned* __restrict__ hmax){
  __shared__ float sWe[HDIM * NHEAD];
  __shared__ unsigned lmax[NHEAD];
  for (int i = threadIdx.x; i < HDIM * NHEAD; i += blockDim.x) sWe[i] = We[i];
  if (threadIdx.x < NHEAD) lmax[threadIdx.x] = 0u;
  __syncthreads();
  int lane = threadIdx.x & 31, wave = threadIdx.x >> 5;
  for (long long ed = (long long)blockIdx.x * 8 + wave; ed < EE; ed += (long long)gridDim.x * 8){
    int s = src[ed], d = dst[ed];
    const float* qp = Xq + (long long)d * HDIM + lane * 4;
    const float* kp = Xk + (long long)s * HDIM + lane * 4;
    const float* ep = efeat + ed * HDIM + lane * 4;
    float p[8];
#pragma unroll
    for (int h = 0; h < 8; ++h) p[h] = 0.f;
    float qk = 0.f;
#pragma unroll
    for (int j = 0; j < 4; ++j){
      float ev = ep[j];
      qk += qp[j] * kp[j];
      int k = lane * 4 + j;
#pragma unroll
      for (int h = 0; h < 8; ++h) p[h] += ev * sWe[k * 8 + h];
    }
    p[lane >> 2] += qk * 0.25f;      // 1/sqrt(D), D=16
#pragma unroll
    for (int off = 16; off >= 1; off >>= 1)
#pragma unroll
      for (int h = 0; h < 8; ++h) p[h] += __shfl_xor(p[h], off, 32);
    if (lane < 8){
      attn[ed * 8 + lane] = p[lane];
      atomicMax(&lmax[lane], fenc(p[lane]));
    }
  }
  __syncthreads();
  if (threadIdx.x < 8) atomicMax(&hmax[threadIdx.x], lmax[threadIdx.x]);
}

// exp(attn - max_h) in place, and denom[dst,h] += w
__global__ void edge_exp(float* __restrict__ attn, const int* __restrict__ dst,
                         float* __restrict__ denom, const unsigned* __restrict__ hmax){
  long long n = EE * 8;
  long long i = (long long)blockIdx.x * blockDim.x + threadIdx.x;
  long long st = (long long)gridDim.x * blockDim.x;
  for (; i < n; i += st){
    int h = (int)(i & 7);
    long long ed = i >> 3;
    float w = expf(attn[i] - fdec(hmax[h]));
    attn[i] = w;
    atomicAdd(&denom[(long long)dst[ed] * 8 + h], w);
  }
}

// agg[dst] += (attn_exp/(denom+1e-9)) * Xv[src]  (wave per edge, 4 dims/lane)
__global__ void edge_agg(const float* __restrict__ attn, const float* __restrict__ denom,
                         const float* __restrict__ Xv, const int* __restrict__ src,
                         const int* __restrict__ dst, float* __restrict__ agg){
  int lane = threadIdx.x & 31, wave = threadIdx.x >> 5;
  long long ed = (long long)blockIdx.x * 8 + wave;
  if (ed >= EE) return;
  int s = src[ed], d = dst[ed];
  int h = lane >> 2;
  float w = attn[ed * 8 + h] / (denom[(long long)d * 8 + h] + 1e-9f);
  const float* vp = Xv + (long long)s * HDIM + lane * 4;
  float* op = agg + (long long)d * HDIM + lane * 4;
#pragma unroll
  for (int j = 0; j < 4; ++j) atomicAdd(&op[j], w * vp[j]);
}

// ---------------- row LayerNorm: out = LN(a+b)*g+beta, wave per 128-row ----------------
__global__ void ln_rows(const float* __restrict__ a, const float* __restrict__ b,
                        const float* __restrict__ g, const float* __restrict__ beta,
                        float* __restrict__ outf, bf16_t* __restrict__ outb, long long M){
  int lane = threadIdx.x & 31, wave = threadIdx.x >> 5;
  long long row = (long long)blockIdx.x * 8 + wave;
  if (row >= M) return;
  int c0 = lane * 4;
  float v[4], s = 0.f, sq = 0.f;
#pragma unroll
  for (int j = 0; j < 4; ++j){
    v[j] = a[row * HDIM + c0 + j] + b[row * HDIM + c0 + j];
    s += v[j]; sq += v[j] * v[j];
  }
#pragma unroll
  for (int off = 16; off >= 1; off >>= 1){
    s  += __shfl_xor(s,  off, 32);
    sq += __shfl_xor(sq, off, 32);
  }
  float mu = s * (1.f / HDIM);
  float var = sq * (1.f / HDIM) - mu * mu;
  float rstd = rsqrtf(var + 1e-5f);
#pragma unroll
  for (int j = 0; j < 4; ++j){
    float o = (v[j] - mu) * rstd * g[c0 + j] + beta[c0 + j];
    outf[row * HDIM + c0 + j] = o;
    if (outb) outb[row * HDIM + c0 + j] = (bf16_t)o;
  }
}

// ---------------- fused edge update: e_out = LN(e + concat(x1[src],x1[dst]) @ W_eu + b_eu) ----------------
// block = 16 edges x 128 outputs; 8 waves, one 16x16 WMMA tile each, K=256
// gathered on the fly; software-pipelined fragment fetch.
__global__ void edge_update(const bf16_t* __restrict__ x1b, const int* __restrict__ src,
                            const int* __restrict__ dst, const bf16_t* __restrict__ WeuT,
                            const float* __restrict__ b_eu, const float* __restrict__ e_in,
                            const float* __restrict__ ge, const float* __restrict__ bee,
                            float* __restrict__ e_out){
  __shared__ float tile[16][HDIM + 4];
  __shared__ float smu[16], srstd[16];
  int lane = threadIdx.x & 31, wave = threadIdx.x >> 5;
  long long m0 = (long long)blockIdx.x * 16;
  int r = lane & 15;
  long long er = m0 + r;
  int s = src[er], d = dst[er];
  const bf16_t* rs = x1b + (long long)s * HDIM;  // k in [0,128)
  const bf16_t* rd = x1b + (long long)d * HDIM;  // k in [128,256)
  int n0 = wave << 4;

  // A fragment loader for the gathered concat(src,dst) row, K = 256
  auto load_a_gather = [&](int k0) -> v16bf {
    int ka = k0 + ((lane & 16) ? 8 : 0);
    int kb = ka + 16;
    union { uint4 u; bf16_t h[8]; } u0, u1;
    u0.u = *(const uint4*)(((ka < 128) ? rs : rd) + (ka & 127));
    u1.u = *(const uint4*)(((kb < 128) ? rs : rd) + (kb & 127));
    v16bf f;
#pragma unroll
    for (int i = 0; i < 8; ++i){ f[i] = u0.h[i]; f[i + 8] = u1.h[i]; }
    return f;
  };

  v8f c = {};
  v16bf a = load_a_gather(0);
  v16bf b = load_b_frag(WeuT, 256, n0, 0, lane);
#pragma unroll
  for (int k0 = 0; k0 < 224; k0 += 32){
    v16bf an = load_a_gather(k0 + 32);
    v16bf bn = load_b_frag(WeuT, 256, n0, k0 + 32, lane);
    c = __builtin_amdgcn_wmma_f32_16x16x32_bf16(false, a, false, b, (short)0, c, false, false);
    a = an; b = bn;
  }
  c = __builtin_amdgcn_wmma_f32_16x16x32_bf16(false, a, false, b, (short)0, c, false, false);

  int col = n0 + r;
  float bi = b_eu[col];
#pragma unroll
  for (int rr = 0; rr < 8; ++rr){
    int row = (lane & 16) ? (8 + rr) : rr;
    tile[row][col] = c[rr] + bi;
  }
  __syncthreads();
  int row2 = threadIdx.x >> 4;
  int c0 = (threadIdx.x & 15) * 8;
  const float* ein = e_in + (m0 + row2) * HDIM + c0;
#pragma unroll
  for (int j = 0; j < 8; ++j) tile[row2][c0 + j] += ein[j];
  __syncthreads();
  if (threadIdx.x < 16){
    float s1 = 0.f, s2 = 0.f;
    for (int cc = 0; cc < HDIM; ++cc){
      float v = tile[threadIdx.x][cc];
      s1 += v; s2 += v * v;
    }
    float mu = s1 * (1.f / HDIM);
    float var = s2 * (1.f / HDIM) - mu * mu;
    smu[threadIdx.x] = mu;
    srstd[threadIdx.x] = rsqrtf(var + 1e-5f);
  }
  __syncthreads();
  float mu = smu[row2], rs2 = srstd[row2];
  float* eo = e_out + (m0 + row2) * HDIM + c0;
#pragma unroll
  for (int j = 0; j < 8; ++j)
    eo[j] = (tile[row2][c0 + j] - mu) * rs2 * ge[c0 + j] + bee[c0 + j];
}

// ---------------- host launch ----------------
extern "C" void kernel_launch(void* const* d_in, const int* in_sizes, int n_in,
                              void* d_out, int out_size, void* d_ws, size_t ws_size,
                              hipStream_t stream){
  const float* x   = (const float*)d_in[0];
  const float* e   = (const float*)d_in[1];
  const int* eidx  = (const int*)d_in[2];
  const int* src   = eidx;
  const int* dst   = eidx + EE;
  const float* Wq  = (const float*)d_in[3];
  const float* Wk  = (const float*)d_in[4];
  const float* Wv  = (const float*)d_in[5];
  const float* We  = (const float*)d_in[6];
  const float* Weu = (const float*)d_in[7];
  const float* beu = (const float*)d_in[8];
  const float* Wout= (const float*)d_in[9];
  const float* bout= (const float*)d_in[10];
  const float* Wf1 = (const float*)d_in[11];
  const float* bf1 = (const float*)d_in[12];
  const float* Wf2 = (const float*)d_in[13];
  const float* bf2 = (const float*)d_in[14];
  const float* g1  = (const float*)d_in[15];
  const float* be1 = (const float*)d_in[16];
  const float* g2  = (const float*)d_in[17];
  const float* be2 = (const float*)d_in[18];
  const float* geW = (const float*)d_in[19];
  const float* bee = (const float*)d_in[20];

  float* xout = (float*)d_out;              // [N,128]
  float* eout = xout + NN * HDIM;           // [E,128]

  // workspace layout (256B aligned)
  char* ws = (char*)d_ws;
  size_t off = 0;
  auto take = [&](size_t bytes) -> char* {
    char* p = ws + off;
    off += (bytes + 255) & ~(size_t)255;
    return p;
  };
  const size_t NBf = (size_t)NN * HDIM * sizeof(float);
  const size_t NBh = (size_t)NN * HDIM * sizeof(bf16_t);
  float*  Xq    = (float*)take(NBf);
  float*  Xk    = (float*)take(NBf);
  float*  Xv    = (float*)take(NBf);
  bf16_t* xb    = (bf16_t*)take(NBh);
  float*  attn  = (float*)take((size_t)EE * 8 * sizeof(float));
  float*  denom = (float*)take((size_t)NN * 8 * sizeof(float));
  unsigned* hmax= (unsigned*)take(8 * sizeof(unsigned));
  float*  agg   = (float*)take(NBf);
  bf16_t* aggb  = (bf16_t*)take(NBh);
  float*  proj  = (float*)take(NBf);
  float*  x1f   = (float*)take(NBf);
  bf16_t* x1b   = (bf16_t*)take(NBh);
  bf16_t* hb    = (bf16_t*)take((size_t)NN * 256 * sizeof(bf16_t));
  float*  ffo   = (float*)take(NBf);
  bf16_t* WqT   = (bf16_t*)take(128 * 128 * sizeof(bf16_t));
  bf16_t* WkT   = (bf16_t*)take(128 * 128 * sizeof(bf16_t));
  bf16_t* WvT   = (bf16_t*)take(128 * 128 * sizeof(bf16_t));
  bf16_t* WoutT = (bf16_t*)take(128 * 128 * sizeof(bf16_t));
  bf16_t* WeuT  = (bf16_t*)take(256 * 128 * sizeof(bf16_t));
  bf16_t* Wf1T  = (bf16_t*)take(128 * 256 * sizeof(bf16_t));
  bf16_t* Wf2T  = (bf16_t*)take(256 * 128 * sizeof(bf16_t));

  // init accumulators
  fill_f32<<<256, 256, 0, stream>>>(denom, NN * 8);
  fill_f32<<<2048, 256, 0, stream>>>(agg, NN * HDIM);
  fill_u32<<<1, 32, 0, stream>>>(hmax, 8);

  // weight prep (transpose + bf16)
  transpose_bf16<<<64, 256, 0, stream>>>(Wq,  WqT,  128, 128);
  transpose_bf16<<<64, 256, 0, stream>>>(Wk,  WkT,  128, 128);
  transpose_bf16<<<64, 256, 0, stream>>>(Wv,  WvT,  128, 128);
  transpose_bf16<<<64, 256, 0, stream>>>(Wout,WoutT,128, 128);
  transpose_bf16<<<128, 256, 0, stream>>>(Weu, WeuT, 256, 128);
  transpose_bf16<<<128, 256, 0, stream>>>(Wf1, Wf1T, 128, 256);
  transpose_bf16<<<128, 256, 0, stream>>>(Wf2, Wf2T, 256, 128);
  cvt_bf16<<<4096, 256, 0, stream>>>(x, xb, NN * HDIM);

  // node-level Q/K/V projections (algebraic hoist of the E-scale GEMMs)
  gemm_bf16<<<NN / 16, 256, 0, stream>>>(xb, WqT, nullptr, Xq, nullptr, (int)NN, 128, 128, 0);
  gemm_bf16<<<NN / 16, 256, 0, stream>>>(xb, WkT, nullptr, Xk, nullptr, (int)NN, 128, 128, 0);
  gemm_bf16<<<NN / 16, 256, 0, stream>>>(xb, WvT, nullptr, Xv, nullptr, (int)NN, 128, 128, 0);

  // attention
  edge_logits<<<2048, 256, 0, stream>>>(Xq, Xk, e, We, src, dst, attn, hmax);
  edge_exp<<<8192, 256, 0, stream>>>(attn, dst, denom, hmax);
  edge_agg<<<EE / 8, 256, 0, stream>>>(attn, denom, Xv, src, dst, agg);

  // output projection + first LN
  cvt_bf16<<<4096, 256, 0, stream>>>(agg, aggb, NN * HDIM);
  gemm_bf16<<<NN / 16, 256, 0, stream>>>(aggb, WoutT, bout, proj, nullptr, (int)NN, 128, 128, 0);
  ln_rows<<<(NN + 7) / 8, 256, 0, stream>>>(x, proj, g1, be1, x1f, x1b, NN);

  // FFN + second LN -> x output
  gemm_bf16<<<NN / 16, 256, 0, stream>>>(x1b, Wf1T, bf1, nullptr, hb, (int)NN, 128, 256, 1);
  gemm_bf16<<<NN / 16, 256, 0, stream>>>(hb, Wf2T, bf2, ffo, nullptr, (int)NN, 256, 128, 0);
  ln_rows<<<(NN + 7) / 8, 256, 0, stream>>>(x1f, ffo, g2, be2, xout, nullptr, NN);

  // fused edge-feature update (the 52 GF GEMM) -> e output
  edge_update<<<EE / 16, 256, 0, stream>>>(x1b, src, dst, WeuT, beu, e, geW, bee, eout);
}
